// MZIBlockLinear_17892833755445
// MI455X (gfx1250) — matrix operand order repaired and checked by reading, not compiled, not run on hardware
//
#include <hip/hip_runtime.h>
#include <math.h>
#include <stdint.h>

// MZI unitary decompose + apply, for MI455X (gfx1250).
//
// Pipeline:
//   1) fp64 CholeskyQR of w (canonical Q with diag(R)>0, matching the
//      reference's sign fix):  G = W^T W ; G = L L^T ; Q = W L^-T
//   2) out = x @ Q^T as a split-f16 WMMA GEMM (hi/lo decomposition,
//      f32 accumulate -> ~fp32 accuracy at f16 matrix-core rates),
//      software-pipelined through LDS using CDNA5 async-to-LDS copies
//      (ASYNCcnt) for the pre-swizzled B fragments.
//
// ws layout: [0,8MB) G fp64 | [8,16MB) X/Q fp64 | [16,18MB) Bhi | [18,20MB) Blo

#define NDIM 1024

typedef _Float16 v16h __attribute__((ext_vector_type(16)));
typedef _Float16 v4h  __attribute__((ext_vector_type(4)));
typedef float    v8f  __attribute__((ext_vector_type(8)));

// ---------------------------------------------------------------- small ops

__global__ void f32_to_f64_kernel(const float* __restrict__ src,
                                  double* __restrict__ dst, int n) {
  int i = blockIdx.x * blockDim.x + threadIdx.x;
  if (i < n) dst[i] = (double)src[i];
}

// G[i][j] = sum_k W[k][i] * W[k][j]  (fp64 accumulate; W is 4MB, L2-resident)
__global__ void gram_f64_kernel(const float* __restrict__ W,
                                double* __restrict__ G) {
  int j = blockIdx.x * 16 + threadIdx.x;
  int i = blockIdx.y * 16 + threadIdx.y;
  double acc = 0.0;
  for (int k = 0; k < NDIM; ++k)
    acc += (double)W[k * NDIM + i] * (double)W[k * NDIM + j];
  G[(size_t)i * NDIM + j] = acc;
}

// Unblocked Cholesky of the 64x64 diagonal block at (kb,kb). One workgroup.
__global__ void potf2_f64_kernel(double* __restrict__ G, int kb) {
  const int tid = threadIdx.x;
  double* A = G + (size_t)kb * NDIM + kb;
  for (int s = 0; s < 64; ++s) {
    if (tid == 0) A[s * NDIM + s] = sqrt(A[s * NDIM + s]);
    __syncthreads();
    double d = A[s * NDIM + s];
    for (int r = s + 1 + tid; r < 64; r += blockDim.x) A[r * NDIM + s] /= d;
    __syncthreads();
    for (int idx = tid; idx < 64 * 64; idx += blockDim.x) {
      int r = idx >> 6, c = idx & 63;
      if (c > s && r >= c)
        A[r * NDIM + c] -= A[r * NDIM + s] * A[c * NDIM + s];
    }
    __syncthreads();
  }
}

// Row-parallel solve: each row r of X (64 columns) solves x * L^T = x
// against a 64x64 lower-triangular L (forward substitution).
__global__ void trsm_rows_f64_kernel(double* __restrict__ X, int rows, int ldx,
                                     const double* __restrict__ L, int ldl) {
  int r = blockIdx.x * blockDim.x + threadIdx.x;
  if (r >= rows) return;
  double* xr = X + (size_t)r * ldx;
  for (int c = 0; c < 64; ++c) {
    double acc = xr[c];
    for (int t = 0; t < c; ++t) acc -= xr[t] * L[(size_t)c * ldl + t];
    xr[c] = acc / L[(size_t)c * ldl + c];
  }
}

// C[r][c] -= sum_k A[r][k] * B[c][k]   (fp64, everything L2-resident)
__global__ void gemm_nt_sub_f64_kernel(double* __restrict__ C,
                                       const double* __restrict__ A,
                                       const double* __restrict__ B,
                                       int Mr, int Nr, int Kr,
                                       int ldc, int lda, int ldb) {
  int c = blockIdx.x * 16 + threadIdx.x;
  int r = blockIdx.y * 16 + threadIdx.y;
  if (r >= Mr || c >= Nr) return;
  double acc = 0.0;
  for (int k = 0; k < Kr; ++k)
    acc += A[(size_t)r * lda + k] * B[(size_t)c * ldb + k];
  C[(size_t)r * ldc + c] -= acc;
}

// ------------------------------------------------- B-fragment preparation
// B = Q^T (B[k][n] = Q[n][k]) split into f16 hi/lo and written directly in
// WMMA 16x16x32 B-fragment order: tile (tk,tn): 32 lanes x 16 halves
// contiguous; lane l holds n = tn*16 + (l&15); element e holds
// k = tk*32 + (l<16 ? e : 16+e)   (contiguous k order; A uses the same).
__global__ void prep_frag_kernel(const double* __restrict__ Q,
                                 _Float16* __restrict__ Bhi,
                                 _Float16* __restrict__ Blo) {
  int idx = blockIdx.x * blockDim.x + threadIdx.x;  // 0 .. 1024*1024-1
  int s    = idx & 15;
  int lane = (idx >> 4) & 31;
  int tile = idx >> 9;       // tk * 64 + tn
  int tn   = tile & 63;
  int tk   = tile >> 6;
  int nn   = tn * 16 + (lane & 15);
  int kk   = tk * 32 + ((lane < 16) ? s : 16 + s);
  float v = (float)Q[(size_t)nn * NDIM + kk];
  _Float16 h = (_Float16)v;
  float res = v - (float)h;
  Bhi[idx] = h;
  Blo[idx] = (_Float16)res;
}

// ------------------------------------------------------------ big WMMA GEMM
// out[32768x1024] = x[32768x1024] * B[1024x1024]
// Workgroup: 256 threads = 8 waves (2 x 4), wg tile 128(M) x 256(N),
// wave tile 64x64 (16 WMMA accumulators). K pipelined in steps of 32 through
// double-buffered LDS:
//   - B hi/lo fragments: contiguous 16KB blocks, copied with
//     global_load_async_to_lds_b128 (ASYNCcnt), issued for step+1 during
//     step's compute, waited with s_wait_asynccnt before the barrier.
//   - A: global float4 loads -> f16 hi/lo split in VGPRs -> ds_store_b64
//     into fragment layout, shared by all 4 n-wave-columns.
__global__ __launch_bounds__(256) void xqt_wmma_kernel(
    const float* __restrict__ X, const _Float16* __restrict__ Bhi,
    const _Float16* __restrict__ Blo, float* __restrict__ Out) {
  const int K = 1024, Nn = 1024, NSTEP = 32;

  __shared__ _Float16 Ahs[2][8 * 512];    //  8KB per buf
  __shared__ _Float16 Als[2][8 * 512];    //  8KB per buf
  __shared__ _Float16 Bhs[2][16 * 512];   // 16KB per buf
  __shared__ _Float16 Bls[2][16 * 512];   // 16KB per buf

  const int tid  = threadIdx.x;
  const int wave = tid >> 5, lane = tid & 31;
  const int half = lane >> 4, lm = lane & 15;
  const int wm = wave >> 2, wn = wave & 3;
  const int nb = blockIdx.x & 3, mb = blockIdx.x >> 2;

  v8f acc[4][4];
#pragma unroll
  for (int i = 0; i < 4; ++i)
#pragma unroll
    for (int j = 0; j < 4; ++j)
#pragma unroll
      for (int r = 0; r < 8; ++r) acc[i][j][r] = 0.0f;

  // ---- stage helpers -------------------------------------------------
  auto issueB = [&](int tk, int buf) {
    // 16 consecutive B tiles = 16KB contiguous per hi/lo array
    const _Float16* sh = Bhi + ((size_t)(tk * 64 + nb * 16)) * 512;
    const _Float16* sl = Blo + ((size_t)(tk * 64 + nb * 16)) * 512;
    unsigned ldh = (unsigned)(uintptr_t)&Bhs[buf][0];
    unsigned ldl = (unsigned)(uintptr_t)&Bls[buf][0];
#pragma unroll
    for (int r = 0; r < 4; ++r) {
      unsigned voff = (unsigned)((tid + r * 256) * 16);  // bytes
      asm volatile("global_load_async_to_lds_b128 %0, %1, %2"
                   :: "v"(ldh + voff), "v"(voff), "s"(sh) : "memory");
      asm volatile("global_load_async_to_lds_b128 %0, %1, %2"
                   :: "v"(ldl + voff), "v"(voff), "s"(sl) : "memory");
    }
  };
  auto loadA = [&](int tk, float4* areg) {
#pragma unroll
    for (int r = 0; r < 4; ++r) {
      int f = tid + r * 256;         // 1024 float4 = 128 rows x 32 k
      int lr = f >> 3;               // local row 0..127
      int kk = (f & 7) * 4;          // local k 0..28
      areg[r] = *(const float4*)(X + (size_t)(mb * 128 + lr) * K + tk * 32 + kk);
    }
  };
  auto storeA = [&](const float4* areg, int buf) {
#pragma unroll
    for (int r = 0; r < 4; ++r) {
      int f = tid + r * 256;
      int lr = f >> 3;
      int kk = (f & 7) * 4;
      int tilei = lr >> 4;
      int lane_s = (lr & 15) + 16 * (kk >> 4);
      int slot = kk & 15;
      int off = tilei * 512 + lane_s * 16 + slot;
      const float* av = (const float*)&areg[r];
      v4h h, l;
#pragma unroll
      for (int s = 0; s < 4; ++s) {
        _Float16 hh = (_Float16)av[s];
        h[s] = hh;
        l[s] = (_Float16)(av[s] - (float)hh);
      }
      *(v4h*)&Ahs[buf][off] = h;
      *(v4h*)&Als[buf][off] = l;
    }
  };

  // ---- prologue: stage 0 into buf 0 ----------------------------------
  {
    issueB(0, 0);
    float4 areg[4];
    loadA(0, areg);
    storeA(areg, 0);
    asm volatile("s_wait_asynccnt 0x0" ::: "memory");
    __syncthreads();
  }

  // ---- main pipelined K loop -----------------------------------------
  for (int step = 0; step < NSTEP; ++step) {
    const int buf = step & 1;
    const bool hasNext = (step + 1) < NSTEP;

    if (hasNext) issueB(step + 1, buf ^ 1);   // async B -> LDS (next)
    float4 areg[4];
    if (hasNext) loadA(step + 1, areg);       // global A (next)

    // compute current stage from LDS
    v16h ah[4], al[4], bh[4], bl[4];
#pragma unroll
    for (int i = 0; i < 4; ++i) {
      ah[i] = *(const v16h*)&Ahs[buf][(wm * 4 + i) * 512 + lane * 16];
      al[i] = *(const v16h*)&Als[buf][(wm * 4 + i) * 512 + lane * 16];
    }
#pragma unroll
    for (int j = 0; j < 4; ++j) {
      bh[j] = *(const v16h*)&Bhs[buf][(wn * 4 + j) * 512 + lane * 16];
      bl[j] = *(const v16h*)&Bls[buf][(wn * 4 + j) * 512 + lane * 16];
    }
#pragma unroll
    for (int i = 0; i < 4; ++i)
#pragma unroll
      for (int j = 0; j < 4; ++j) {
        acc[i][j] = __builtin_amdgcn_wmma_f32_16x16x32_f16(
            false, ah[i], false, bh[j], (short)0, acc[i][j], false, false);
        acc[i][j] = __builtin_amdgcn_wmma_f32_16x16x32_f16(
            false, ah[i], false, bl[j], (short)0, acc[i][j], false, false);
        acc[i][j] = __builtin_amdgcn_wmma_f32_16x16x32_f16(
            false, al[i], false, bh[j], (short)0, acc[i][j], false, false);
      }

    if (hasNext) {
      storeA(areg, buf ^ 1);                  // hi/lo split -> LDS (next)
      asm volatile("s_wait_asynccnt 0x0" ::: "memory");
    }
    __syncthreads();
  }

  // ---- store C: VGPR r, lane -> m = r + 8*half, n = lm ---------------
  const int m0 = mb * 128 + wm * 64;
  const int n0 = nb * 256 + wn * 64;
#pragma unroll
  for (int i = 0; i < 4; ++i)
#pragma unroll
    for (int j = 0; j < 4; ++j)
#pragma unroll
      for (int r = 0; r < 8; ++r) {
        int m = m0 + i * 16 + r + 8 * half;
        int nn = n0 + j * 16 + lm;
        __builtin_nontemporal_store(acc[i][j][r], Out + (size_t)m * Nn + nn);
      }
}

// ------------------------------------------------------------------- launch

extern "C" void kernel_launch(void* const* d_in, const int* in_sizes, int n_in,
                              void* d_out, int out_size, void* d_ws, size_t ws_size,
                              hipStream_t stream) {
  const float* x = (const float*)d_in[0];   // [8,4096,1024] fp32
  const float* w = (const float*)d_in[1];   // [1024,1024]  fp32
  float* out = (float*)d_out;               // [8,4096,1024] fp32
  char* ws = (char*)d_ws;                   // needs 20 MB

  double*    G   = (double*)(ws);                        // 8 MB
  double*    X64 = (double*)(ws + (size_t)(8 << 20));    // 8 MB (W -> Q)
  _Float16*  Bhi = (_Float16*)(ws + (size_t)(16 << 20)); // 2 MB
  _Float16*  Blo = (_Float16*)(ws + (size_t)(18 << 20)); // 2 MB

  const int n = NDIM, nel = NDIM * NDIM;

  // X64 = (double)W
  f32_to_f64_kernel<<<(nel + 255) / 256, 256, 0, stream>>>(w, X64, nel);
  // G = W^T W (fp64)
  gram_f64_kernel<<<dim3(64, 64), dim3(16, 16), 0, stream>>>(w, G);

  // Blocked Cholesky: G = L L^T (L in lower triangle of G), b = 64
  for (int kb = 0; kb < n; kb += 64) {
    potf2_f64_kernel<<<1, 256, 0, stream>>>(G, kb);
    int rem = n - kb - 64;
    if (rem > 0) {
      trsm_rows_f64_kernel<<<(rem + 63) / 64, 64, 0, stream>>>(
          G + (size_t)(kb + 64) * n + kb, rem, n,
          G + (size_t)kb * n + kb, n);
      gemm_nt_sub_f64_kernel<<<dim3((rem + 15) / 16, (rem + 15) / 16),
                               dim3(16, 16), 0, stream>>>(
          G + (size_t)(kb + 64) * n + (kb + 64),
          G + (size_t)(kb + 64) * n + kb,
          G + (size_t)(kb + 64) * n + kb, rem, rem, 64, n, n, n);
    }
  }

  // Blocked TRSM: X64 <- X64 * L^-T  (i.e. Q = W L^-T), in place
  for (int jb = 0; jb < n; jb += 64) {
    if (jb > 0) {
      gemm_nt_sub_f64_kernel<<<dim3(4, 64), dim3(16, 16), 0, stream>>>(
          X64 + jb, X64, G + (size_t)jb * n, n, 64, jb, n, n, n);
    }
    trsm_rows_f64_kernel<<<(n + 63) / 64, 64, 0, stream>>>(
        X64 + jb, n, n, G + (size_t)jb * n + jb, n);
  }

  // Split Q^T into f16 hi/lo WMMA B-fragments
  prep_frag_kernel<<<(nel + 255) / 256, 256, 0, stream>>>(X64, Bhi, Blo);

  // out = x @ Q^T : 256 x 4 blocks of 128x256, 8 waves each
  xqt_wmma_kernel<<<1024, 256, 0, stream>>>(x, Bhi, Blo, out);
}
